// CausalSelfAttention_33595234189360
// MI455X (gfx1250) — matrix operand order repaired and checked by reading, not compiled, not run on hardware
//
#include <hip/hip_runtime.h>
#include <hip/hip_bf16.h>

typedef __attribute__((ext_vector_type(16))) _Float16 v16h;
typedef __attribute__((ext_vector_type(8)))  _Float16 v8h;
typedef __attribute__((ext_vector_type(8)))  float    v8f;
typedef __attribute__((ext_vector_type(4)))  int      v4i;

// Optional CDNA5 async global->LDS path (guarded so compile can't regress).
#if defined(__has_builtin)
#if __has_builtin(__builtin_amdgcn_global_load_async_to_lds_b128) && \
    __has_builtin(__builtin_amdgcn_s_wait_asynccnt)
#define USE_ASYNC_LDS 1
#endif
#endif

#if defined(USE_ASYNC_LDS)
// Probe-derived signature: (v4i addrspace(1)*, v4i addrspace(3)*, imm offset, imm cpol)
typedef __attribute__((address_space(1))) v4i g1v4i;
typedef __attribute__((address_space(3))) v4i s3v4i;
static __device__ inline void async_copy16(const _Float16* gsrc, _Float16* ldst) {
  __builtin_amdgcn_global_load_async_to_lds_b128((g1v4i*)gsrc, (s3v4i*)ldst, 0, 0);
}
#endif

// ---------------------------------------------------------------------------
// WMMA fragment helpers (wave32, v_wmma_f32_16x16x32_f16)
// A-matrix 16x32 f16 layout (ISA 7.12.2): lane m = lane&15, half = lane>>4,
//   elements 0..7  -> K =  8*half + e
//   elements 8..15 -> K = 16 + 8*half + (e-8)
// B-matrix 32x16 f16 (N-major storage tile[n][k]): lane n = lane&15,
//   elements e -> same K as A (8*half + e, then 16 + 8*half + (e-8))
// C/D 16x16 f32: element r, lane L -> M = r + 8*(L>>4), N = L&15
// ---------------------------------------------------------------------------
static __device__ inline v16h make_frag(const _Float16* p0, const _Float16* p1) {
  v8h lo = *(const v8h*)p0;
  v8h hi = *(const v8h*)p1;
  v16h r;
#pragma unroll
  for (int i = 0; i < 8; ++i) { r[i] = lo[i]; r[i + 8] = hi[i]; }
  return r;
}

static __device__ inline v16h load_a_frag(const _Float16* tile, int ld) {
  const int lane = threadIdx.x & 31;
  const int m = lane & 15, half = lane >> 4;
  const _Float16* row = tile + m * ld;
  return make_frag(row + 8 * half, row + 16 + 8 * half);
}

static __device__ inline v16h load_b_frag(const _Float16* tile, int ld) {
  const int lane = threadIdx.x & 31;
  const int n = lane & 15, half = lane >> 4;
  const _Float16* row = tile + n * ld;
  return make_frag(row + 8 * half, row + 16 + 8 * half);
}

// ---------------------------------------------------------------------------
// Generic GEMM: C[M,N] = A[M,K] fp32 @ B[K,N] fp32, fp32 out, f16 WMMA inside.
// Block tile 128x128x32, 256 threads (8 waves), wave tile 32x64 (2x4 WMMA).
// Register-staged double-buffered LDS: next tile's global loads are issued
// before the WMMA block so HBM latency hides behind the matrix pipe.
// M%128==0, N%128==0, K%32==0 guaranteed by problem sizes.
// ---------------------------------------------------------------------------
#define GLD 40  // LDS row stride in f16 (80B: 16B-aligned fragment chunks)

__global__ __launch_bounds__(256) void gemm_f32_wmma(
    const float* __restrict__ A, const float* __restrict__ B,
    float* __restrict__ C, int M, int N, int K) {
  __shared__ __align__(16) _Float16 As[2][128 * GLD];
  __shared__ __align__(16) _Float16 Bs[2][128 * GLD];

  const int tid = threadIdx.x;
  const int wid = tid >> 5;
  const int m0 = blockIdx.y * 128;
  const int n0 = blockIdx.x * 128;
  const int wm = (wid & 3) * 32;
  const int wn = (wid >> 2) * 64;
  const int ar = tid >> 3, ac = (tid & 7) * 4;   // A-tile slot: 128r x 32c
  const int br = tid >> 5, bc = (tid & 31) * 4;  // B-tile slot: 32r x 128c
  (void)M;

  v8f zero8 = {};
  v8f acc[2][4];
#pragma unroll
  for (int s = 0; s < 2; ++s)
#pragma unroll
    for (int u = 0; u < 4; ++u) acc[s][u] = zero8;

  float4 ra[4], rb[4];
  auto gload = [&](int k0) {
#pragma unroll
    for (int i = 0; i < 4; ++i)
      ra[i] = *(const float4*)(A + (size_t)(m0 + ar + i * 32) * K + k0 + ac);
#pragma unroll
    for (int i = 0; i < 4; ++i)
      rb[i] = *(const float4*)(B + (size_t)(k0 + br + i * 8) * N + n0 + bc);
  };
  auto stash = [&](int buf) {
#pragma unroll
    for (int i = 0; i < 4; ++i) {
      _Float16* dst = &As[buf][(ar + i * 32) * GLD + ac];
      dst[0] = (_Float16)ra[i].x; dst[1] = (_Float16)ra[i].y;
      dst[2] = (_Float16)ra[i].z; dst[3] = (_Float16)ra[i].w;
    }
#pragma unroll
    for (int i = 0; i < 4; ++i) {
      _Float16* dst = &Bs[buf][bc * GLD + br + i * 8];
      dst[0 * GLD] = (_Float16)rb[i].x;
      dst[1 * GLD] = (_Float16)rb[i].y;
      dst[2 * GLD] = (_Float16)rb[i].z;
      dst[3 * GLD] = (_Float16)rb[i].w;
    }
  };

  const int nk = K / 32;
  gload(0);
  stash(0);
  int buf = 0;
  for (int kt = 0; kt < nk; ++kt) {
    __syncthreads();
    if (kt + 1 < nk) gload((kt + 1) * 32);  // in-flight during WMMA below

    v16h a[2], b[4];
#pragma unroll
    for (int s = 0; s < 2; ++s)
      a[s] = load_a_frag(&As[buf][(wm + 16 * s) * GLD], GLD);
#pragma unroll
    for (int u = 0; u < 4; ++u)
      b[u] = load_b_frag(&Bs[buf][(wn + 16 * u) * GLD], GLD);
#pragma unroll
    for (int s = 0; s < 2; ++s)
#pragma unroll
      for (int u = 0; u < 4; ++u)
        acc[s][u] = __builtin_amdgcn_wmma_f32_16x16x32_f16(
            false, a[s], false, b[u], (short)0, acc[s][u], false, false);

    if (kt + 1 < nk) stash(buf ^ 1);  // waits loadcnt here, after WMMAs issued
    buf ^= 1;
  }

  const int lane = tid & 31;
  const int nlo = lane & 15, half = lane >> 4;
#pragma unroll
  for (int s = 0; s < 2; ++s)
#pragma unroll
    for (int u = 0; u < 4; ++u)
#pragma unroll
      for (int r = 0; r < 8; ++r) {
        int row = m0 + wm + 16 * s + r + 8 * half;
        int col = n0 + wn + 16 * u + nlo;
        C[(size_t)row * N + col] = acc[s][u][r];
      }
}

// ---------------------------------------------------------------------------
// RoPE + RMS-norm on Q,K (fp32 -> f16), plus V fp32 -> f16 conversion.
// One block per token; 24 waves: 16 Q heads, 4 K heads, 4 V heads.
// ---------------------------------------------------------------------------
__global__ __launch_bounds__(768) void rope_rms_cvt(
    const float* __restrict__ Qf, const float* __restrict__ Kf,
    const float* __restrict__ Vf, const float* __restrict__ cosT,
    const float* __restrict__ sinT, _Float16* __restrict__ Qh,
    _Float16* __restrict__ Kh, _Float16* __restrict__ Vh, int T) {
  const int n = blockIdx.x;          // token index in [0, B*T)
  const int t = n % T;
  const int wid = threadIdx.x >> 5;  // 0..23
  const int lane = threadIdx.x & 31;

  if (wid < 20) {
    const float* src;
    _Float16* dst;
    if (wid < 16) {
      src = Qf + (size_t)n * 2048 + wid * 128;
      dst = Qh + (size_t)n * 2048 + wid * 128;
    } else {
      src = Kf + (size_t)n * 512 + (wid - 16) * 128;
      dst = Kh + (size_t)n * 512 + (wid - 16) * 128;
    }
    const int j = lane * 2;  // 0..62, rope pairs (j, j+64)
    float x1a = src[j], x1b = src[j + 1];
    float x2a = src[j + 64], x2b = src[j + 65];
    float ca = cosT[(size_t)t * 64 + j], cb = cosT[(size_t)t * 64 + j + 1];
    float sa = sinT[(size_t)t * 64 + j], sb = sinT[(size_t)t * 64 + j + 1];
    float y1a = x1a * ca - x2a * sa, y1b = x1b * cb - x2b * sb;
    float y2a = x1a * sa + x2a * ca, y2b = x1b * sb + x2b * cb;
    float ss = y1a * y1a + y1b * y1b + y2a * y2a + y2b * y2b;
#pragma unroll
    for (int m = 16; m >= 1; m >>= 1) ss += __shfl_xor(ss, m, 32);
    const float inv = rsqrtf(ss * (1.0f / 128.0f) + 1e-6f);
    dst[j] = (_Float16)(y1a * inv);
    dst[j + 1] = (_Float16)(y1b * inv);
    dst[j + 64] = (_Float16)(y2a * inv);
    dst[j + 65] = (_Float16)(y2b * inv);
  } else {
    const int h = wid - 20;
    const float4 f = *(const float4*)(Vf + (size_t)n * 512 + h * 128 + lane * 4);
    _Float16* dst = Vh + (size_t)n * 512 + h * 128 + lane * 4;
    dst[0] = (_Float16)f.x; dst[1] = (_Float16)f.y;
    dst[2] = (_Float16)f.z; dst[3] = (_Float16)f.w;
  }
}

// ---------------------------------------------------------------------------
// Flash attention (causal, GQA rep=4). One block per (b, h, 64-row q tile),
// 4 waves, each wave owns 16 q rows. Online softmax, WMMA for QK^T and PV.
// Q/K tiles copied global->LDS via async-tensor path when available.
// ---------------------------------------------------------------------------
#define QK_LD 136  // 64x128 tile row stride (f16); 272B rows, 16B aligned chunks
#define V_LD 72    // 128x64 / 16x64 tile row stride; 144B rows

__global__ __launch_bounds__(128) void flash_attn(
    const _Float16* __restrict__ Qh, const _Float16* __restrict__ Kh,
    const _Float16* __restrict__ Vh, float* __restrict__ O, int T) {
  __shared__ __align__(16) _Float16 Qs[64 * QK_LD];
  __shared__ __align__(16) _Float16 Ks[64 * QK_LD];
  __shared__ __align__(16) _Float16 Vs[128 * V_LD];  // transposed: [dim][key]
  __shared__ __align__(16) _Float16 Ps[4][16 * V_LD];

  const int bh = blockIdx.x;
  const int b = bh >> 4, h = bh & 15, kvh = h >> 2;
  const int qt = blockIdx.y;
  const int tid = threadIdx.x, w = tid >> 5, lane = tid & 31;
  const int nlo = lane & 15, half = lane >> 4;
  const float scale = 0.08838834764831845f;  // 1/sqrt(128)

  // Load Q tile (64 x 128 f16)
  const size_t qbase = ((size_t)(b * T + qt * 64)) * 2048 + h * 128;
#pragma unroll
  for (int i = 0; i < 8; ++i) {
    int idx = tid + i * 128;  // 0..1023 = 64 rows * 16 chunks of 8 halves
    int r = idx >> 4, c8 = (idx & 15) * 8;
    const _Float16* g = Qh + qbase + (size_t)r * 2048 + c8;
    _Float16* l = Qs + r * QK_LD + c8;
#if defined(USE_ASYNC_LDS)
    async_copy16(g, l);
#else
    *(v8h*)l = *(const v8h*)g;
#endif
  }
#if defined(USE_ASYNC_LDS)
  __builtin_amdgcn_s_wait_asynccnt(0);
#endif
  __syncthreads();

  // Pre-load Q A-fragments for this wave's 16 rows (4 k-steps), reused all tiles
  v16h aq[4];
#pragma unroll
  for (int ks = 0; ks < 4; ++ks)
    aq[ks] = load_a_frag(Qs + (w * 16) * QK_LD + ks * 32, QK_LD);

  v8f zero8 = {};
  v8f acco[8];
#pragma unroll
  for (int i = 0; i < 8; ++i) acco[i] = zero8;
  float mrow[8], lrow[8];
#pragma unroll
  for (int r = 0; r < 8; ++r) { mrow[r] = -3.0e38f; lrow[r] = 0.f; }

  const size_t kvbase = ((size_t)(b * T)) * 512 + kvh * 128;

  for (int kt = 0; kt <= qt; ++kt) {
    // Load K tile (row-major, async if available); V transposed via registers
#pragma unroll
    for (int i = 0; i < 8; ++i) {
      int idx = tid + i * 128;
      int r = idx >> 4, c8 = (idx & 15) * 8;
      const size_t src = kvbase + (size_t)(kt * 64 + r) * 512 + c8;
#if defined(USE_ASYNC_LDS)
      async_copy16(Kh + src, Ks + r * QK_LD + c8);
#else
      *(v8h*)(Ks + r * QK_LD + c8) = *(const v8h*)(Kh + src);
#endif
      v8h v = *(const v8h*)(Vh + src);
#pragma unroll
      for (int j = 0; j < 8; ++j) Vs[(c8 + j) * V_LD + r] = v[j];
    }
    // Prefetch next K/V tile into cache hierarchy
    if (kt < qt) {
      const size_t nsrc = kvbase + (size_t)((kt + 1) * 64 + (tid >> 1)) * 512 +
                          (size_t)(tid & 1) * 64;
      __builtin_prefetch(Kh + nsrc, 0, 1);
      __builtin_prefetch(Vh + nsrc, 0, 1);
    }
#if defined(USE_ASYNC_LDS)
    __builtin_amdgcn_s_wait_asynccnt(0);
#endif
    __syncthreads();

    // S = scale * (Q K^T), 16 q rows x 64 keys per wave
    v8f s[4];
#pragma unroll
    for (int u = 0; u < 4; ++u) {
      v8f a = zero8;
#pragma unroll
      for (int ks = 0; ks < 4; ++ks) {
        v16h bk = load_b_frag(Ks + (u * 16) * QK_LD + ks * 32, QK_LD);
        a = __builtin_amdgcn_wmma_f32_16x16x32_f16(false, aq[ks], false, bk,
                                                   (short)0, a, false, false);
      }
      s[u] = a;
    }

    const int qrow0 = qt * 64 + w * 16 + 8 * half;
    const bool diag = (kt == qt);
#pragma unroll
    for (int u = 0; u < 4; ++u) {
      int kc = kt * 64 + u * 16 + nlo;
#pragma unroll
      for (int r = 0; r < 8; ++r) {
        float v = s[u][r] * scale;
        if (diag && kc > qrow0 + r) v = -3.0e38f;
        s[u][r] = v;
      }
    }

    // Online softmax per row (cross-lane reduce within the 16-lane half)
#pragma unroll
    for (int r = 0; r < 8; ++r) {
      float mx = s[0][r];
#pragma unroll
      for (int u = 1; u < 4; ++u) mx = fmaxf(mx, s[u][r]);
#pragma unroll
      for (int m = 8; m >= 1; m >>= 1) mx = fmaxf(mx, __shfl_xor(mx, m, 32));
      float mnew = fmaxf(mrow[r], mx);
      float corr = __expf(mrow[r] - mnew);
      float rs = 0.f;
#pragma unroll
      for (int u = 0; u < 4; ++u) {
        float p = __expf(s[u][r] - mnew);
        s[u][r] = p;
        rs += p;
      }
#pragma unroll
      for (int m = 8; m >= 1; m >>= 1) rs += __shfl_xor(rs, m, 32);
      lrow[r] = lrow[r] * corr + rs;
      mrow[r] = mnew;
#pragma unroll
      for (int nt = 0; nt < 8; ++nt) acco[nt][r] *= corr;
    }

    // Stage P (C/D layout -> A layout) through per-wave LDS
    _Float16* ps = &Ps[w][0];
#pragma unroll
    for (int u = 0; u < 4; ++u)
#pragma unroll
      for (int r = 0; r < 8; ++r)
        ps[(r + 8 * half) * V_LD + u * 16 + nlo] = (_Float16)s[u][r];
    asm volatile("s_wait_dscnt 0" ::: "memory");

    // O += P V
#pragma unroll
    for (int ks = 0; ks < 2; ++ks) {
      v16h ap = load_a_frag(ps + ks * 32, V_LD);
#pragma unroll
      for (int nt = 0; nt < 8; ++nt) {
        v16h bv = load_b_frag(Vs + (nt * 16) * V_LD + ks * 32, V_LD);
        acco[nt] = __builtin_amdgcn_wmma_f32_16x16x32_f16(
            false, ap, false, bv, (short)0, acco[nt], false, false);
      }
    }
    __syncthreads();
  }

  // Normalize and write fp32 attention output [n][h*128+dim]
  const int row_l = qt * 64 + w * 16 + 8 * half;
#pragma unroll
  for (int r = 0; r < 8; ++r) {
    float inv = 1.0f / lrow[r];
    size_t rowbase = ((size_t)(b * T + row_l + r)) * 2048 + h * 128;
#pragma unroll
    for (int nt = 0; nt < 8; ++nt)
      O[rowbase + nt * 16 + nlo] = acco[nt][r] * inv;
  }
}

// ---------------------------------------------------------------------------
extern "C" void kernel_launch(void* const* d_in, const int* in_sizes, int n_in,
                              void* d_out, int out_size, void* d_ws,
                              size_t ws_size, hipStream_t stream) {
  (void)n_in; (void)out_size; (void)ws_size;
  const float* x = (const float*)d_in[0];
  const float* cosT = (const float*)d_in[1];
  const float* sinT = (const float*)d_in[2];
  const float* Wq = (const float*)d_in[3];
  const float* Wk = (const float*)d_in[4];
  const float* Wv = (const float*)d_in[5];
  const float* Wo = (const float*)d_in[6];
  float* out = (float*)d_out;

  const int T = in_sizes[1] / 64;    // cos: T x (HEAD_DIM/2)
  const int C = in_sizes[3] / 2048;  // Wq: C x (16*128)
  const int BT = in_sizes[0] / C;
  const int B = BT / T;

  char* ws = (char*)d_ws;
  const size_t szQf = (size_t)BT * 2048 * sizeof(float);
  const size_t szKf = (size_t)BT * 512 * sizeof(float);
  float* Qf = (float*)ws;
  float* Kf = (float*)(ws + szQf);
  float* Vf = (float*)(ws + szQf + szKf);
  _Float16* Qhp = (_Float16*)(ws + szQf + 2 * szKf);
  _Float16* Khp = Qhp + (size_t)BT * 2048;
  _Float16* Vhp = Khp + (size_t)BT * 512;
  float* attO = (float*)ws;  // reuse Qf region after f16 conversion

  dim3 blk(256);
  gemm_f32_wmma<<<dim3(2048 / 128, BT / 128), blk, 0, stream>>>(x, Wq, Qf, BT, 2048, C);
  gemm_f32_wmma<<<dim3(512 / 128, BT / 128), blk, 0, stream>>>(x, Wk, Kf, BT, 512, C);
  gemm_f32_wmma<<<dim3(512 / 128, BT / 128), blk, 0, stream>>>(x, Wv, Vf, BT, 512, C);
  rope_rms_cvt<<<BT, 768, 0, stream>>>(Qf, Kf, Vf, cosT, sinT, Qhp, Khp, Vhp, T);
  flash_attn<<<dim3(B * 16, T / 64), dim3(128), 0, stream>>>(Qhp, Khp, Vhp, attO, T);
  gemm_f32_wmma<<<dim3(C / 128, BT / 128), blk, 0, stream>>>(attO, Wo, out, BT, C, 2048);
}
